// EnhancedMultiHeadSelfAttention_68040871903542
// MI455X (gfx1250) — matrix operand order repaired and checked by reading, not compiled, not run on hardware
//
#include <hip/hip_runtime.h>
#include <cstdint>
#include <cstddef>

// ---------------------------------------------------------------------------
// Sizes (compile-time constants for this problem)
// ---------------------------------------------------------------------------
#define L_SEQ   2048
#define DMODEL  1024
#define NHEADS  16
#define HEADDIM 64
#define DFF     4096

// ---------------------------------------------------------------------------
// Types / helpers
// ---------------------------------------------------------------------------
typedef __attribute__((ext_vector_type(16))) __bf16 v16bf;
typedef __attribute__((ext_vector_type(8)))  float  v8f;

struct U32x8 { uint4 lo, hi; };

__device__ __forceinline__ v16bf make_frag(uint4 lo, uint4 hi) {
  U32x8 t; t.lo = lo; t.hi = hi;
  return __builtin_bit_cast(v16bf, t);
}

// fp32 -> bf16 bits, round-to-nearest-even
__device__ __forceinline__ unsigned short f2bfu(float f) {
  uint32_t u = __builtin_bit_cast(uint32_t, f);
  uint32_t r = (u + 0x7FFFu + ((u >> 16) & 1u)) >> 16;
  return (unsigned short)r;
}
__device__ __forceinline__ uint32_t pack_bf2(float a, float b) {
  return (uint32_t)f2bfu(a) | ((uint32_t)f2bfu(b) << 16);
}

// Generic __shared__ pointer -> 32-bit LDS byte address (generic LDS
// addresses are {shared_aperture, lds_offset}; low 32 bits are the offset).
__device__ __forceinline__ uint32_t lds_addr32(const void* p) {
  return (uint32_t)(uintptr_t)p;
}

// CDNA5 async global->LDS DMA (ASYNCcnt-tracked), per-lane 16B.
__device__ __forceinline__ void async_load_b128(uint32_t lds_addr, const void* g) {
  asm volatile("global_load_async_to_lds_b128 %0, %1, off"
               :: "v"(lds_addr), "v"(g) : "memory");
}
__device__ __forceinline__ void wait_asynccnt0() {
  asm volatile("s_wait_asynccnt 0x0" ::: "memory");
}

// ---------------------------------------------------------------------------
// Weight transpose + fp32->bf16 cast:  in[K][N] f32  ->  out[N][K] bf16
// (Bt layout: WMMA B-fragment lane n reads contiguous K)
// ---------------------------------------------------------------------------
__global__ void transpose_cast_kernel(const float* __restrict__ in,
                                      unsigned short* __restrict__ out,
                                      int K, int N) {
  __shared__ float tile[32][33];
  int tx = threadIdx.x & 31;
  int ty = threadIdx.x >> 5;          // 0..7
  int n0 = blockIdx.x * 32;
  int k0 = blockIdx.y * 32;
#pragma unroll
  for (int i = 0; i < 4; ++i) {
    int k = ty + i * 8;
    tile[k][tx] = in[(size_t)(k0 + k) * N + n0 + tx];
  }
  __syncthreads();
#pragma unroll
  for (int i = 0; i < 4; ++i) {
    int n = ty + i * 8;
    out[(size_t)(n0 + n) * K + k0 + tx] = f2bfu(tile[tx][n]);
  }
}

// ---------------------------------------------------------------------------
// LayerNorm over D=1024, output bf16. One block (256 thr) per row.
// ---------------------------------------------------------------------------
__global__ void layernorm_kernel(const float* __restrict__ x,
                                 const float* __restrict__ g,
                                 const float* __restrict__ b,
                                 unsigned short* __restrict__ out) {
  int row = blockIdx.x;
  int tid = threadIdx.x;
  int lane = tid & 31, wave = tid >> 5;
  float4 v = ((const float4*)(x + (size_t)row * DMODEL))[tid];
  float s  = v.x + v.y + v.z + v.w;
  float s2 = v.x * v.x + v.y * v.y + v.z * v.z + v.w * v.w;
#pragma unroll
  for (int m = 1; m <= 16; m <<= 1) {
    s  += __shfl_xor(s,  m, 32);
    s2 += __shfl_xor(s2, m, 32);
  }
  __shared__ float red[16];
  if (lane == 0) { red[wave] = s; red[8 + wave] = s2; }
  __syncthreads();
  float ts = 0.f, ts2 = 0.f;
#pragma unroll
  for (int i = 0; i < 8; ++i) { ts += red[i]; ts2 += red[8 + i]; }
  float mu   = ts * (1.0f / DMODEL);
  float var  = ts2 * (1.0f / DMODEL) - mu * mu;
  float rstd = rsqrtf(var + 1e-5f);
  float4 gv = ((const float4*)g)[tid];
  float4 bv = ((const float4*)b)[tid];
  float n0 = (v.x - mu) * rstd * gv.x + bv.x;
  float n1 = (v.y - mu) * rstd * gv.y + bv.y;
  float n2 = (v.z - mu) * rstd * gv.z + bv.z;
  float n3 = (v.w - mu) * rstd * gv.w + bv.w;
  uint2 p; p.x = pack_bf2(n0, n1); p.y = pack_bf2(n2, n3);
  ((uint2*)(out + (size_t)row * DMODEL))[tid] = p;
}

// ---------------------------------------------------------------------------
// Split qkv[L][3][H][64] fp32 -> per-head bf16 buffers with cosine normalize:
//   qb[H][L][64]  = q / max(||q||,1e-12) * 0.125
//   kb[H][L][64]  = k / max(||k||,1e-12)
//   vt[H][64][L]  = v transposed (Bt layout for P@V)
// ---------------------------------------------------------------------------
__global__ void qkv_split_norm_kernel(const float* __restrict__ qkv,
                                      unsigned short* __restrict__ qb,
                                      unsigned short* __restrict__ kb,
                                      unsigned short* __restrict__ vt) {
  int l = blockIdx.x;
  int wave = threadIdx.x >> 5, lane = threadIdx.x & 31;
  int h = blockIdx.y * 8 + wave;
  int d = lane * 2;
  const float* base = qkv + (size_t)l * 3 * DMODEL + h * HEADDIM + d;
  float2 q = *(const float2*)(base);
  float2 k = *(const float2*)(base + DMODEL);
  float2 v = *(const float2*)(base + 2 * DMODEL);
  float qs = q.x * q.x + q.y * q.y;
  float ks = k.x * k.x + k.y * k.y;
#pragma unroll
  for (int m = 1; m <= 16; m <<= 1) {
    qs += __shfl_xor(qs, m, 32);
    ks += __shfl_xor(ks, m, 32);
  }
  float qf = 0.125f / fmaxf(sqrtf(qs), 1e-12f);   // scaling = HEAD_DIM^-0.5
  float kf = 1.0f   / fmaxf(sqrtf(ks), 1e-12f);
  size_t o = ((size_t)h * L_SEQ + l) * HEADDIM + d;
  ((uint32_t*)qb)[o >> 1] = pack_bf2(q.x * qf, q.y * qf);
  ((uint32_t*)kb)[o >> 1] = pack_bf2(k.x * kf, k.y * kf);
  vt[((size_t)h * HEADDIM + d    ) * L_SEQ + l] = f2bfu(v.x);
  vt[((size_t)h * HEADDIM + d + 1) * L_SEQ + l] = f2bfu(v.y);
}

// ---------------------------------------------------------------------------
// bf16 WMMA GEMM: C[M][N] = A[M][K] @ Bt[N][K]^T + bias, templated epilogue.
// Block tile 128x128, BK=32, 8 waves in 2x4; wave tile 64x32 (8 wmma/step).
// Double-buffered LDS filled by async global->LDS DMA (ASYNCcnt), one
// barrier per K-step; next tile's DMA overlaps current tile's WMMAs.
// ---------------------------------------------------------------------------
#define EPI_F32       0
#define EPI_RES_F32   2
#define EPI_GELU_BF16 3

template <int EPI>
__global__ __launch_bounds__(256) void gemm_bf16_kernel(
    const unsigned short* __restrict__ A, const unsigned short* __restrict__ Bt,
    const float* __restrict__ bias, const float* __restrict__ res,
    float* __restrict__ outF, unsigned short* __restrict__ outH,
    int M, int N, int K) {
  constexpr int BM = 128, BN = 128, BK = 32, LD = 40;  // LD padded vs 32
  __shared__ unsigned short sA[2][BM * LD];
  __shared__ unsigned short sB[2][BN * LD];
  int tid  = threadIdx.x;
  int lane = tid & 31, wave = tid >> 5;
  int hl = lane >> 4, lm = lane & 15;
  int wm = wave >> 2, wn = wave & 3;
  int rowBase = blockIdx.y * BM, colBase = blockIdx.x * BN;

  int ldRow = tid >> 1;           // 128 rows, 2 threads/row
  int ldOff = (tid & 1) * 16;     // 16 halves each

  v8f acc[4][2];
#pragma unroll
  for (int i = 0; i < 4; ++i)
#pragma unroll
    for (int j = 0; j < 2; ++j)
#pragma unroll
      for (int r = 0; r < 8; ++r) acc[i][j][r] = 0.f;

  const unsigned short* gA = A  + (size_t)(rowBase + ldRow) * K + ldOff;
  const unsigned short* gB = Bt + (size_t)(colBase + ldRow) * K + ldOff;

  // per-thread LDS destination byte addresses for both buffers
  uint32_t dA[2], dB[2];
#pragma unroll
  for (int bf = 0; bf < 2; ++bf) {
    dA[bf] = lds_addr32(&sA[bf][ldRow * LD + ldOff]);
    dB[bf] = lds_addr32(&sB[bf][ldRow * LD + ldOff]);
  }

  // issue one 128x32 A tile + 128x32 B tile as 4 async 16B DMAs per thread
  auto issue_tile = [&](int bf, int k0) {
    async_load_b128(dA[bf],      gA + k0);
    async_load_b128(dA[bf] + 16, gA + k0 + 8);
    async_load_b128(dB[bf],      gB + k0);
    async_load_b128(dB[bf] + 16, gB + k0 + 8);
  };

  issue_tile(0, 0);  // prologue fill

  for (int k0 = 0; k0 < K; k0 += BK) {
    int cur = (k0 >> 5) & 1;
    wait_asynccnt0();     // my DMAs into buf[cur] have landed
    __syncthreads();      // everyone's have
    if (k0 + BK < K) issue_tile(1 - cur, k0 + BK);    // overlap next fill
    if (k0 + 4 * BK < K) {                            // keep L2 warm
      __builtin_prefetch(gA + k0 + 4 * BK, 0, 0);
      __builtin_prefetch(gB + k0 + 4 * BK, 0, 0);
    }

    v16bf af[4], bfv[2];
#pragma unroll
    for (int mt = 0; mt < 4; ++mt) {        // A frag: lane m = row, K swizzled
      int r = wm * 64 + mt * 16 + lm;
      uint4 lo = *(const uint4*)&sA[cur][r * LD + hl * 8];
      uint4 hi = *(const uint4*)&sA[cur][r * LD + hl * 8 + 16];
      af[mt] = make_frag(lo, hi);
    }
#pragma unroll
    for (int nt = 0; nt < 2; ++nt) {        // B frag: lane n = col, K swizzled
      int r = wn * 32 + nt * 16 + lm;
      uint4 lo = *(const uint4*)&sB[cur][r * LD + hl * 8];
      uint4 hi = *(const uint4*)&sB[cur][r * LD + hl * 8 + 16];
      bfv[nt] = make_frag(lo, hi);
    }
#pragma unroll
    for (int mt = 0; mt < 4; ++mt)
#pragma unroll
      for (int nt = 0; nt < 2; ++nt)
        acc[mt][nt] = __builtin_amdgcn_wmma_f32_16x16x32_bf16(
            false, af[mt], false, bfv[nt], (short)0, acc[mt][nt], false, false);
  }

#pragma unroll
  for (int nt = 0; nt < 2; ++nt) {
    int col = colBase + wn * 32 + nt * 16 + lm;
    float bv = bias[col];
#pragma unroll
    for (int mt = 0; mt < 4; ++mt) {
#pragma unroll
      for (int r = 0; r < 8; ++r) {
        int row = rowBase + wm * 64 + mt * 16 + hl * 8 + r;
        float v = acc[mt][nt][r] + bv;
        size_t idx = (size_t)row * N + col;
        if (EPI == EPI_F32) {
          outF[idx] = v;
        } else if (EPI == EPI_RES_F32) {
          outF[idx] = v + res[idx];
        } else {  // exact GELU -> bf16
          float gl = 0.5f * v * (1.0f + erff(v * 0.70710678118654752f));
          outH[idx] = f2bfu(gl);
        }
      }
    }
  }
}

// ---------------------------------------------------------------------------
// Flash attention. Block: 128 query rows of one head (8 waves x 16 rows).
// Streams 32-key blocks: S = q@k^T (4 wmma) -> bias/clip -> online softmax
// (shfl_xor within the C-layout 16-lane halves) -> P rotated C->A layout via
// per-wave LDS scratch -> O += P@V (4 wmma). Output bf16 attn[L][1024].
// ---------------------------------------------------------------------------
__global__ __launch_bounds__(256) void flash_attn_kernel(
    const unsigned short* __restrict__ Q,   // [H][L][64]
    const unsigned short* __restrict__ Kb,  // [H][L][64]   (Bt for q@k^T)
    const unsigned short* __restrict__ Vt,  // [H][64][L]   (Bt for p@v)
    const float* __restrict__ lcc,
    unsigned short* __restrict__ attn) {    // [L][1024] bf16
  constexpr int LDP = 40;
  __shared__ unsigned short sP[8 * 16 * LDP];
  int tid = threadIdx.x;
  int lane = tid & 31, wave = tid >> 5;
  int hl = lane >> 4, lm = lane & 15;
  int h = blockIdx.y;
  int qbase = blockIdx.x * 128 + wave * 16;
  unsigned short* myP = &sP[wave * 16 * LDP];

  v16bf qf[2];
  {
    const unsigned short* qrow = Q + ((size_t)h * L_SEQ + qbase + lm) * HEADDIM;
#pragma unroll
    for (int c = 0; c < 2; ++c) {
      uint4 lo = *(const uint4*)(qrow + c * 32 + hl * 8);
      uint4 hi = *(const uint4*)(qrow + c * 32 + hl * 8 + 16);
      qf[c] = make_frag(lo, hi);
    }
  }
  float lq[8];
#pragma unroll
  for (int r = 0; r < 8; ++r) lq[r] = lcc[qbase + hl * 8 + r] * 0.05f;

  float mrun[8], lrun[8];
#pragma unroll
  for (int r = 0; r < 8; ++r) { mrun[r] = -1e30f; lrun[r] = 0.f; }
  v8f o[4];
#pragma unroll
  for (int t = 0; t < 4; ++t)
#pragma unroll
    for (int r = 0; r < 8; ++r) o[t][r] = 0.f;

  for (int kb = 0; kb < L_SEQ; kb += 32) {
    v16bf kf0[2], kf1[2];
    {
      const unsigned short* kr0 = Kb + ((size_t)h * L_SEQ + kb + lm) * HEADDIM;
      const unsigned short* kr1 = kr0 + 16 * HEADDIM;
#pragma unroll
      for (int c = 0; c < 2; ++c) {
        kf0[c] = make_frag(*(const uint4*)(kr0 + c * 32 + hl * 8),
                           *(const uint4*)(kr0 + c * 32 + hl * 8 + 16));
        kf1[c] = make_frag(*(const uint4*)(kr1 + c * 32 + hl * 8),
                           *(const uint4*)(kr1 + c * 32 + hl * 8 + 16));
      }
    }
    v8f s0, s1;
#pragma unroll
    for (int r = 0; r < 8; ++r) { s0[r] = 0.f; s1[r] = 0.f; }
    s0 = __builtin_amdgcn_wmma_f32_16x16x32_bf16(false, qf[0], false, kf0[0], (short)0, s0, false, false);
    s0 = __builtin_amdgcn_wmma_f32_16x16x32_bf16(false, qf[1], false, kf0[1], (short)0, s0, false, false);
    s1 = __builtin_amdgcn_wmma_f32_16x16x32_bf16(false, qf[0], false, kf1[0], (short)0, s1, false, false);
    s1 = __builtin_amdgcn_wmma_f32_16x16x32_bf16(false, qf[1], false, kf1[1], (short)0, s1, false, false);

    float lk0 = lcc[kb + lm] * 0.05f;
    float lk1 = lcc[kb + 16 + lm] * 0.05f;
    float rmax[8];
#pragma unroll
    for (int r = 0; r < 8; ++r) {
      float a = fminf(fmaxf(s0[r] + lq[r] + lk0, -10.f), 10.f);
      float b = fminf(fmaxf(s1[r] + lq[r] + lk1, -10.f), 10.f);
      s0[r] = a; s1[r] = b;
      rmax[r] = fmaxf(a, b);
    }
#pragma unroll
    for (int m = 1; m <= 8; m <<= 1)
#pragma unroll
      for (int r = 0; r < 8; ++r)
        rmax[r] = fmaxf(rmax[r], __shfl_xor(rmax[r], m, 32));

    float rs[8];
#pragma unroll
    for (int r = 0; r < 8; ++r) {
      float mn   = fmaxf(mrun[r], rmax[r]);
      float corr = __expf(mrun[r] - mn);
      mrun[r] = mn;
      float p0 = __expf(s0[r] - mn);
      float p1 = __expf(s1[r] - mn);
      rs[r] = p0 + p1;
      lrun[r] *= corr;
#pragma unroll
      for (int t = 0; t < 4; ++t) o[t][r] *= corr;
      myP[(hl * 8 + r) * LDP + lm]      = f2bfu(p0);
      myP[(hl * 8 + r) * LDP + 16 + lm] = f2bfu(p1);
    }
#pragma unroll
    for (int m = 1; m <= 8; m <<= 1)
#pragma unroll
      for (int r = 0; r < 8; ++r) rs[r] += __shfl_xor(rs[r], m, 32);
#pragma unroll
    for (int r = 0; r < 8; ++r) lrun[r] += rs[r];

    // P (C-layout) -> A-fragment via LDS (in-wave, DScnt ordered)
    v16bf pf = make_frag(*(const uint4*)&myP[lm * LDP + hl * 8],
                         *(const uint4*)&myP[lm * LDP + hl * 8 + 16]);
#pragma unroll
    for (int t = 0; t < 4; ++t) {
      const unsigned short* vrow =
          Vt + ((size_t)h * HEADDIM + t * 16 + lm) * L_SEQ + kb + hl * 8;
      v16bf vf = make_frag(*(const uint4*)vrow, *(const uint4*)(vrow + 16));
      o[t] = __builtin_amdgcn_wmma_f32_16x16x32_bf16(false, pf, false, vf, (short)0, o[t], false, false);
    }
  }

  float inv[8];
#pragma unroll
  for (int r = 0; r < 8; ++r) inv[r] = 1.f / lrun[r];
#pragma unroll
  for (int t = 0; t < 4; ++t)
#pragma unroll
    for (int r = 0; r < 8; ++r)
      attn[(size_t)(qbase + hl * 8 + r) * DMODEL + h * HEADDIM + t * 16 + lm] =
          f2bfu(o[t][r] * inv[r]);
}

// ---------------------------------------------------------------------------
// Host launch
// ---------------------------------------------------------------------------
extern "C" void kernel_launch(void* const* d_in, const int* in_sizes, int n_in,
                              void* d_out, int out_size, void* d_ws, size_t ws_size,
                              hipStream_t stream) {
  (void)in_sizes; (void)n_in; (void)out_size; (void)ws_size;
  const float* x     = (const float*)d_in[0];
  const float* lcc   = (const float*)d_in[1];
  const float* w_qkv = (const float*)d_in[2];
  const float* b_qkv = (const float*)d_in[3];
  const float* w_out = (const float*)d_in[4];
  const float* b_out = (const float*)d_in[5];
  const float* ln1_g = (const float*)d_in[6];
  const float* ln1_b = (const float*)d_in[7];
  const float* ln2_g = (const float*)d_in[8];
  const float* ln2_b = (const float*)d_in[9];
  const float* w_ff1 = (const float*)d_in[10];
  const float* b_ff1 = (const float*)d_in[11];
  const float* w_ff2 = (const float*)d_in[12];
  const float* b_ff2 = (const float*)d_in[13];
  float* out = (float*)d_out;

  char* wsp = (char*)d_ws;
  size_t off = 0;
  auto take = [&](size_t bytes) -> void* {
    void* p = wsp + off;
    off = (off + bytes + 255) & ~(size_t)255;
    return p;
  };
  unsigned short* normed1 = (unsigned short*)take((size_t)L_SEQ * DMODEL * 2);
  unsigned short* wqkvT   = (unsigned short*)take((size_t)3 * DMODEL * DMODEL * 2);
  float*          qkv     = (float*)take((size_t)L_SEQ * 3 * DMODEL * 4);
  unsigned short* qb      = (unsigned short*)take((size_t)NHEADS * L_SEQ * HEADDIM * 2);
  unsigned short* kb      = (unsigned short*)take((size_t)NHEADS * L_SEQ * HEADDIM * 2);
  unsigned short* vt      = (unsigned short*)take((size_t)NHEADS * HEADDIM * L_SEQ * 2);
  unsigned short* attnb   = (unsigned short*)take((size_t)L_SEQ * DMODEL * 2);
  unsigned short* woutT   = (unsigned short*)take((size_t)DMODEL * DMODEL * 2);
  float*          x1      = (float*)take((size_t)L_SEQ * DMODEL * 4);
  unsigned short* normed2 = (unsigned short*)take((size_t)L_SEQ * DMODEL * 2);
  unsigned short* wff1T   = (unsigned short*)take((size_t)DFF * DMODEL * 2);
  unsigned short* hbuf    = (unsigned short*)take((size_t)L_SEQ * DFF * 2);
  unsigned short* wff2T   = (unsigned short*)take((size_t)DMODEL * DFF * 2);

  dim3 blk(256);

  // Weight transpose + bf16 cast (read once; L2-resident afterwards)
  transpose_cast_kernel<<<dim3(3072 / 32, 1024 / 32), blk, 0, stream>>>(w_qkv, wqkvT, 1024, 3072);
  transpose_cast_kernel<<<dim3(1024 / 32, 1024 / 32), blk, 0, stream>>>(w_out, woutT, 1024, 1024);
  transpose_cast_kernel<<<dim3(4096 / 32, 1024 / 32), blk, 0, stream>>>(w_ff1, wff1T, 1024, 4096);
  transpose_cast_kernel<<<dim3(1024 / 32, 4096 / 32), blk, 0, stream>>>(w_ff2, wff2T, 4096, 1024);

  // LN1 -> QKV GEMM -> split/cosine-normalize
  layernorm_kernel<<<dim3(L_SEQ), blk, 0, stream>>>(x, ln1_g, ln1_b, normed1);
  gemm_bf16_kernel<EPI_F32><<<dim3(3072 / 128, L_SEQ / 128), blk, 0, stream>>>(
      normed1, wqkvT, b_qkv, nullptr, qkv, nullptr, L_SEQ, 3072, 1024);
  qkv_split_norm_kernel<<<dim3(L_SEQ, 2), blk, 0, stream>>>(qkv, qb, kb, vt);

  // Streaming (flash) attention
  flash_attn_kernel<<<dim3(L_SEQ / 128, NHEADS), blk, 0, stream>>>(qb, kb, vt, lcc, attnb);

  // Out-proj + residual, LN2, FF1+GELU, FF2 + residual -> d_out
  gemm_bf16_kernel<EPI_RES_F32><<<dim3(1024 / 128, L_SEQ / 128), blk, 0, stream>>>(
      attnb, woutT, b_out, x, x1, nullptr, L_SEQ, 1024, 1024);
  layernorm_kernel<<<dim3(L_SEQ), blk, 0, stream>>>(x1, ln2_g, ln2_b, normed2);
  gemm_bf16_kernel<EPI_GELU_BF16><<<dim3(4096 / 128, L_SEQ / 128), blk, 0, stream>>>(
      normed2, wff1T, b_ff1, nullptr, nullptr, hbuf, L_SEQ, 4096, 1024);
  gemm_bf16_kernel<EPI_RES_F32><<<dim3(1024 / 128, L_SEQ / 128), blk, 0, stream>>>(
      hbuf, wff2T, b_ff2, x1, out, nullptr, L_SEQ, 1024, 4096);
}
